// MultiHeadAttn_17703855194621
// MI455X (gfx1250) — matrix operand order repaired and checked by reading, not compiled, not run on hardware
//
#include <hip/hip_runtime.h>
#include <hip/hip_bf16.h>

typedef __attribute__((ext_vector_type(16))) __bf16 bf16x16;
typedef __attribute__((ext_vector_type(8)))  __bf16 bf16x8;
typedef __attribute__((ext_vector_type(2)))  __bf16 bf16x2;
typedef __attribute__((ext_vector_type(8)))  float  f32x8;

#define B_  4
#define L_  2048
#define D_  1024
#define DK_ 64

#if defined(__has_builtin)
#if __has_builtin(__builtin_amdgcn_cvt_pk_bf16_f32)
#define HAVE_PK_BF16 1
#endif
#endif

// single convert: round-half-away; compiler folds the >>16 into d16_hi stores
static __device__ __forceinline__ __bf16 f2bf(float f) {
    unsigned u = (__builtin_bit_cast(unsigned, f) + 0x8000u) >> 16;
    return __builtin_bit_cast(__bf16, (unsigned short)u);
}

// packed convert of a float pair -> one dword (elem0 = a in [15:0], elem1 = b in [31:16]),
// matching WMMA fragment packing. 2x v_add + 1x v_perm_b32 (or 1 op with hw cvt).
static __device__ __forceinline__ unsigned pk2bf(float a, float b) {
#ifdef HAVE_PK_BF16
    bf16x2 t = __builtin_amdgcn_cvt_pk_bf16_f32(a, b);
    return __builtin_bit_cast(unsigned, t);
#else
    unsigned ua = __builtin_bit_cast(unsigned, a) + 0x8000u;
    unsigned ub = __builtin_bit_cast(unsigned, b) + 0x8000u;
    // dst = { ub[31:16], ua[31:16] } : sel bytes (lsb..msb) = ua.b2, ua.b3, ub.b2, ub.b3
    return __builtin_amdgcn_perm(ub, ua, 0x07060302u);
#endif
}

// fragment assembled at dword / chunk granularity (no element-wise b16 merges)
union frag16 {
    bf16x16  v;
    unsigned u[8];
    bf16x8   h[2];
};

// ---- prep: WT[n][d] = bf16(W[d][n] * scale)   (N-major for WMMA B-fragments)
__global__ void wt_prep(const float* __restrict__ W, __bf16* __restrict__ WT, float scale) {
    int idx = blockIdx.x * blockDim.x + threadIdx.x;   // 65536
    int d = idx & (D_ - 1);
    int n = idx >> 10;
    WT[(size_t)n * D_ + d] = f2bf(W[(size_t)d * DK_ + n] * scale);
}

// ---- prep: WoeT[d][k] = bf16(sum_h Wo[h*64+k][d])  (folded + N-major)
__global__ void wo_fold(const float* __restrict__ Wo, __bf16* __restrict__ woeT) {
    int idx = blockIdx.x * blockDim.x + threadIdx.x;   // 65536
    int k = idx & (DK_ - 1);
    int d = idx >> 6;
    float s = 0.f;
#pragma unroll
    for (int h = 0; h < 16; ++h) s += Wo[(size_t)(h * DK_ + k) * D_ + d];
    woeT[(size_t)d * DK_ + k] = f2bf(s);
}

// ---- shared GEMM body: acc[4] (16x64 tile) = X[m0..m0+15][0..1023] @ W
static __device__ __forceinline__ void proj_body(const float* __restrict__ X,
                                                 const __bf16* __restrict__ WT,
                                                 int m0, int lc, int hi, f32x8 acc[4]) {
    const float* xrow = X + (size_t)(m0 + lc) * D_;
    for (int kk = 0; kk < D_; kk += 32) {
        f32x8 lo = *(const f32x8*)(xrow + kk + hi * 16);
        f32x8 hp = *(const f32x8*)(xrow + kk + hi * 16 + 8);
        frag16 A;
#pragma unroll
        for (int i = 0; i < 4; ++i) {
            A.u[i]     = pk2bf(lo[2 * i], lo[2 * i + 1]);
            A.u[i + 4] = pk2bf(hp[2 * i], hp[2 * i + 1]);
        }
#pragma unroll
        for (int t = 0; t < 4; ++t) {
            bf16x16 b = *(const bf16x16*)(WT + (size_t)(16 * t + lc) * D_ + kk + hi * 16);
            acc[t] = __builtin_amdgcn_wmma_f32_16x16x32_bf16(false, A.v, false, b,
                                                             (short)0, acc[t], false, false);
        }
    }
}

// ---- projection, row-major store: Out[8192][64]
__global__ __launch_bounds__(128) void proj_nn(const float* __restrict__ X,
                                               const __bf16* __restrict__ WT,
                                               __bf16* __restrict__ Out) {
    const int lane = threadIdx.x & 31;
    const int lc = lane & 15, hi = lane >> 4;
    const int m0 = ((blockIdx.x * blockDim.x + threadIdx.x) >> 5) * 16;
    f32x8 acc[4] = {};
    proj_body(X, WT, m0, lc, hi, acc);
#pragma unroll
    for (int t = 0; t < 4; ++t)
#pragma unroll
        for (int r = 0; r < 8; ++r)
            Out[(size_t)(m0 + r + 8 * hi) * DK_ + 16 * t + lc] = f2bf(acc[t][r]);
}

// ---- projection, transposed store: Out[B][64][L]
__global__ __launch_bounds__(128) void proj_nt(const float* __restrict__ X,
                                               const __bf16* __restrict__ WT,
                                               __bf16* __restrict__ Out) {
    const int lane = threadIdx.x & 31;
    const int lc = lane & 15, hi = lane >> 4;
    const int m0 = ((blockIdx.x * blockDim.x + threadIdx.x) >> 5) * 16;
    f32x8 acc[4] = {};
    proj_body(X, WT, m0, lc, hi, acc);
#pragma unroll
    for (int t = 0; t < 4; ++t)
#pragma unroll
        for (int r = 0; r < 8; ++r) {
            int row = m0 + r + 8 * hi;
            int bb = row >> 11, pos = row & (L_ - 1);
            Out[((size_t)bb * DK_ + 16 * t + lc) * L_ + pos] = f2bf(acc[t][r]);
        }
}

// ---- fused flash attention (S^T trick) + output projection via folded Wo
__global__ __launch_bounds__(128) void attn_kernel(const __bf16* __restrict__ qbf,
                                                   const __bf16* __restrict__ kbf,
                                                   const __bf16* __restrict__ vT,
                                                   const int* __restrict__ mask,
                                                   const __bf16* __restrict__ woeT,
                                                   float* __restrict__ out) {
    const int lane = threadIdx.x & 31;
    const int lc = lane & 15, hi = lane >> 4;
    const int w = threadIdx.x >> 5;
    const int tile = (blockIdx.x * blockDim.x + threadIdx.x) >> 5;  // 0..511
    const int b = tile >> 7;
    const int q0 = (tile & 127) * 16;

    __shared__ __align__(16) __bf16 lds_o[4][16][72];

    const __bf16* qrow  = qbf + ((size_t)b * L_ + q0 + lc) * DK_;
    const __bf16* kbase = kbf + (size_t)b * L_ * DK_;
    const __bf16* vbase = vT + (size_t)b * DK_ * L_;
    const int*    mbase = mask + (size_t)b * L_ * L_ + (size_t)(q0 + lc) * L_;

    // loop-invariant q B-fragments (K = dk halves)
    bf16x16 bq0 = *(const bf16x16*)(qrow + hi * 16);
    bf16x16 bq1 = *(const bf16x16*)(qrow + 32 + hi * 16);

    f32x8 o[4] = {};
    float m_cur = -__builtin_huge_valf();
    float l_cur = 0.f;

    for (int k0 = 0; k0 < L_; k0 += 32) {
        if (k0 + 32 < L_) {   // uniform; prefetch next key/value/mask tiles
            __builtin_prefetch(mbase + k0 + 32, 0, 1);
            __builtin_prefetch(kbase + (size_t)(k0 + 32 + lc) * DK_, 0, 1);
            __builtin_prefetch(vbase + (size_t)lc * L_ + k0 + 32, 0, 1);
        }
        f32x8 s[2];
#pragma unroll
        for (int j = 0; j < 2; ++j) {
            const __bf16* krow = kbase + (size_t)(k0 + 16 * j + lc) * DK_;
            frag16 ak0, ak1;
            ak0.h[0] = *(const bf16x8*)(krow + hi * 16);
            ak0.h[1] = *(const bf16x8*)(krow + hi * 16 + 8);
            ak1.h[0] = *(const bf16x8*)(krow + 32 + hi * 16);
            ak1.h[1] = *(const bf16x8*)(krow + 40 + hi * 16);
            f32x8 z = {};
            z = __builtin_amdgcn_wmma_f32_16x16x32_bf16(false, ak0.v, false, bq0,
                                                        (short)0, z, false, false);
            z = __builtin_amdgcn_wmma_f32_16x16x32_bf16(false, ak1.v, false, bq1,
                                                        (short)0, z, false, false);
            const int* mp = mbase + k0 + 16 * j + 8 * hi;
            int4 mlo = *(const int4*)(mp);
            int4 mhi = *(const int4*)(mp + 4);
            int mm[8] = {mlo.x, mlo.y, mlo.z, mlo.w, mhi.x, mhi.y, mhi.z, mhi.w};
#pragma unroll
            for (int r = 0; r < 8; ++r)
                if (mm[r] == 0) z[r] = -__builtin_huge_valf();
            s[j] = z;
        }
        // online softmax: per-lane query = q0 + lc, keys spread over r and hi-half
        float tmax = s[0][0];
#pragma unroll
        for (int r = 0; r < 8; ++r) {
            tmax = fmaxf(tmax, s[0][r]);
            tmax = fmaxf(tmax, s[1][r]);
        }
        tmax = fmaxf(tmax, __shfl_xor(tmax, 16, 32));
        float m_new = fmaxf(m_cur, tmax);
        float scl = __expf(m_cur - m_new);
        float p0[8], p1[8];
        float rsum = 0.f;
#pragma unroll
        for (int r = 0; r < 8; ++r) {
            p0[r] = __expf(s[0][r] - m_new);
            p1[r] = __expf(s[1][r] - m_new);
            rsum += p0[r] + p1[r];
        }
        frag16 PA;   // P 16x32 A-fragment falls out of the S^T C-layout for free
#pragma unroll
        for (int i = 0; i < 4; ++i) {
            PA.u[i]     = pk2bf(p0[2 * i], p0[2 * i + 1]);
            PA.u[i + 4] = pk2bf(p1[2 * i], p1[2 * i + 1]);
        }
        rsum += __shfl_xor(rsum, 16, 32);
        l_cur = l_cur * scl + rsum;
        m_cur = m_new;
        // rescale accumulator rows (O rows = r + 8*hi -> scale lives in lane 8*hi+r)
#pragma unroll
        for (int r = 0; r < 8; ++r) {
            float sr = __shfl(scl, 8 * hi + r, 32);
#pragma unroll
            for (int t = 0; t < 4; ++t) o[t][r] *= sr;
        }
        // O += P @ V   (B-fragments contiguous from vT)
#pragma unroll
        for (int t = 0; t < 4; ++t) {
            frag16 bv;
            bv.h[0] = *(const bf16x8*)(vbase + (size_t)(16 * t + lc) * L_ + k0 + hi * 8);
            bv.h[1] = *(const bf16x8*)(vbase + (size_t)(16 * t + lc) * L_ + k0 + 16 + hi * 8);
            o[t] = __builtin_amdgcn_wmma_f32_16x16x32_bf16(false, PA.v, false, bv.v,
                                                           (short)0, o[t], false, false);
        }
    }

    // normalize and transpose O (C-layout -> A-layout) through per-wave LDS tile
    float linv = 1.f / l_cur;
#pragma unroll
    for (int r = 0; r < 8; ++r) {
        float lr = __shfl(linv, 8 * hi + r, 32);
#pragma unroll
        for (int t = 0; t < 4; ++t)
            lds_o[w][r + 8 * hi][16 * t + lc] = f2bf(o[t][r] * lr);
    }
    const __bf16* orow = &lds_o[w][lc][0];
    frag16 ao0, ao1;
    ao0.h[0] = *(const bf16x8*)(orow + hi * 16);
    ao0.h[1] = *(const bf16x8*)(orow + hi * 16 + 8);
    ao1.h[0] = *(const bf16x8*)(orow + 32 + hi * 16);
    ao1.h[1] = *(const bf16x8*)(orow + 40 + hi * 16);

    // fused epilogue: out[16 x 1024] = O @ Wo_eff
    float* obase = out + ((size_t)b * L_ + q0) * D_;
    for (int dt = 0; dt < 64; ++dt) {
        const __bf16* wrow = woeT + (size_t)(dt * 16 + lc) * DK_;
        bf16x16 bw0 = *(const bf16x16*)(wrow + hi * 16);
        bf16x16 bw1 = *(const bf16x16*)(wrow + 32 + hi * 16);
        f32x8 c = {};
        c = __builtin_amdgcn_wmma_f32_16x16x32_bf16(false, ao0.v, false, bw0,
                                                    (short)0, c, false, false);
        c = __builtin_amdgcn_wmma_f32_16x16x32_bf16(false, ao1.v, false, bw1,
                                                    (short)0, c, false, false);
#pragma unroll
        for (int r = 0; r < 8; ++r)
            obase[(size_t)(r + 8 * hi) * D_ + dt * 16 + lc] = c[r];
    }
}

extern "C" void kernel_launch(void* const* d_in, const int* in_sizes, int n_in,
                              void* d_out, int out_size, void* d_ws, size_t ws_size,
                              hipStream_t stream) {
    const float* query = (const float*)d_in[0];
    const float* key   = (const float*)d_in[1];
    const float* value = (const float*)d_in[2];
    const int*   mask  = (const int*)d_in[3];
    const float* Wq    = (const float*)d_in[4];
    const float* Wk    = (const float*)d_in[5];
    const float* Wv    = (const float*)d_in[6];
    const float* Wo    = (const float*)d_in[7];
    float* out = (float*)d_out;

    char* ws = (char*)d_ws;
    __bf16* qbf  = (__bf16*)(ws);                               // 1 MiB
    __bf16* kbf  = (__bf16*)(ws + (1u << 20));                  // 1 MiB
    __bf16* vT   = (__bf16*)(ws + (2u << 20));                  // 1 MiB
    __bf16* WqT  = (__bf16*)(ws + (3u << 20));                  // 128 KiB each
    __bf16* WkT  = (__bf16*)(ws + (3u << 20) + (128u << 10));
    __bf16* WvT  = (__bf16*)(ws + (3u << 20) + (256u << 10));
    __bf16* woeT = (__bf16*)(ws + (3u << 20) + (384u << 10));

    wt_prep<<<256, 256, 0, stream>>>(Wq, WqT, 0.125f);  // fold 1/sqrt(64) into Wq
    wt_prep<<<256, 256, 0, stream>>>(Wk, WkT, 1.0f);
    wt_prep<<<256, 256, 0, stream>>>(Wv, WvT, 1.0f);
    wo_fold<<<256, 256, 0, stream>>>(Wo, woeT);

    proj_nn<<<128, 128, 0, stream>>>(query, WqT, qbf);   // 512 waves each
    proj_nn<<<128, 128, 0, stream>>>(key,   WkT, kbf);
    proj_nt<<<128, 128, 0, stream>>>(value, WvT, vT);    // store transposed

    attn_kernel<<<128, 128, 0, stream>>>(qbf, kbf, vT, mask, woeT, out);
}
